// DecoderConvWithBuffer_74457553044047
// MI455X (gfx1250) — matrix operand order repaired and checked by reading, not compiled, not run on hardware
//
#include <hip/hip_runtime.h>
#include <math.h>

// ---------------- types for CDNA5 WMMA ----------------
typedef __bf16 bf16;
typedef bf16           v16bf __attribute__((ext_vector_type(16)));
typedef float          v8f   __attribute__((ext_vector_type(8)));
typedef unsigned short us8v  __attribute__((ext_vector_type(8)));
typedef unsigned short us16v __attribute__((ext_vector_type(16)));

struct Us8Pair { us8v lo; us8v hi; };
union  AFrag   { v16bf v; Us8Pair p; };
union  BFrag   { v16bf v; us16v  u; };

// ---------------- problem dimensions ----------------
#define NBLK      32
#define BLK       256
#define NTHREADS  (NBLK * BLK)
#define NWAVES    (NTHREADS / 32)

#define B_        16
#define T_DEC     100
#define T_ENC_    128
#define IN_DIM_   256
#define HIDDEN_   256
#define OUT_DIM_  400
#define MEM_DIM_  80
#define BUF_LEN_  17
#define BUF_DIM_  336         // 256 + 80
#define BUF_SIZE_ 5712        // 17*336
#define KPAD      5728        // 179*32  (padded K for buf GEMMs)
#define NCH       179         // K chunks of 32 for K=5712(+16 pad)
#define H1V       571         // BUF_SIZE/10
#define H1P       576         // padded to 36*16
#define HLK       4352        // 17*256 (K of out GEMM; 136*32 exactly)
#define KG3       7           // split-K groups for the big bp_w1 GEMM

#define OUTS_ELEMS (B_ * T_DEC * OUT_DIM_)   // 640000; attns follow

// ---------------- bf16 helpers ----------------
__device__ __forceinline__ unsigned short f2bf(float x) {
  union { float f; unsigned u; } v; v.f = x;
  unsigned r = v.u + 0x7FFFu + ((v.u >> 16) & 1u);   // round-to-nearest-even
  return (unsigned short)(r >> 16);
}

__device__ __forceinline__ v8f wmma_bf16(v16bf a, v16bf b, v8f c) {
  return __builtin_amdgcn_wmma_f32_16x16x32_bf16(false, a, false, b, (short)0, c,
                                                 false, false);
}

// A fragment (16x32, row-major activations, stride lda in elements, 16B aligned)
__device__ __forceinline__ v16bf load_a_tile(const unsigned short* Ab, int lda, int lane) {
  int m = lane & 15, half = (lane >> 4) << 4;
  const unsigned short* r = Ab + m * lda + half;
  AFrag f;
  f.p.lo = *(const us8v*)(r);
  f.p.hi = *(const us8v*)(r + 8);
  return f.v;
}

// B fragment from fragment-major weight storage: one 32B load per lane.
__device__ __forceinline__ v16bf load_b_frag(const unsigned short* tile, int lane) {
  BFrag f;
  f.u = *(const us16v*)(tile + lane * 16);
  return f.v;
}

// C/D layout: lane l holds col=l&15, rows r + 8*(l>>4) for vgpr r=0..7.
__device__ __forceinline__ void store_partial(float* dst, int ld, v8f acc, int lane) {
  int col = lane & 15, rb = (lane >> 4) << 3;
#pragma unroll
  for (int r = 0; r < 8; ++r) dst[(r + rb) * ld + col] = acc[r];
}

// Rotated software pipeline: preload pair 0; each iteration prefetches pair
// p+1 while computing pair p (two independent accumulators; fixed combine
// order -> deterministic). WMMAs consume fragments loaded one iteration ago,
// so they retire at loadcnt<=4 with the next pair still in flight.
__device__ __forceinline__ v8f gemm_run(const unsigned short* __restrict__ Ab, int lda,
                                        const unsigned short* __restrict__ Bb,
                                        int bstride, int n, int lane) {
  v8f acc0 = {}, acc1 = {};
  if (n >= 2) {
    v16bf a0 = load_a_tile(Ab, lda, lane);
    v16bf b0 = load_b_frag(Bb, lane);
    v16bf a1 = load_a_tile(Ab + 32, lda, lane);
    v16bf b1 = load_b_frag(Bb + bstride, lane);
    int c = 2;
    for (; c + 2 <= n; c += 2) {
      v16bf a2 = load_a_tile(Ab + c * 32, lda, lane);
      v16bf b2 = load_b_frag(Bb + c * bstride, lane);
      v16bf a3 = load_a_tile(Ab + (c + 1) * 32, lda, lane);
      v16bf b3 = load_b_frag(Bb + (c + 1) * bstride, lane);
      acc0 = wmma_bf16(a0, b0, acc0);
      acc1 = wmma_bf16(a1, b1, acc1);
      a0 = a2; b0 = b2; a1 = a3; b1 = b3;
    }
    if (c < n) {                                // odd tail: prefetch, then drain
      v16bf a2 = load_a_tile(Ab + c * 32, lda, lane);
      v16bf b2 = load_b_frag(Bb + c * bstride, lane);
      acc0 = wmma_bf16(a0, b0, acc0);
      acc1 = wmma_bf16(a1, b1, acc1);
      acc0 = wmma_bf16(a2, b2, acc0);
    } else {
      acc0 = wmma_bf16(a0, b0, acc0);
      acc1 = wmma_bf16(a1, b1, acc1);
    }
  } else if (n == 1) {
    acc0 = wmma_bf16(load_a_tile(Ab, lda, lane), load_b_frag(Bb, lane), acc0);
  }
#pragma unroll
  for (int r = 0; r < 8; ++r) acc0[r] += acc1[r];
  return acc0;
}

// P5 special A: rows are (b,l) pairs inside padded flat buffer, K guarded to 336.
__device__ __forceinline__ v16bf load_a_h(const unsigned short* zb, int mt, int kbase,
                                          int lane) {
  int m = lane & 15, half = (lane >> 4) << 4;
  int rg = mt * 16 + m;
  int b = rg / 17, l = rg - b * 17;
  const unsigned short* row = zb + b * KPAD + l * BUF_DIM_;
  us8v z = {};
  AFrag f;
  int k0 = kbase + half, k1 = kbase + half + 8;
  f.p.lo = (k0 < BUF_DIM_) ? *(const us8v*)(row + k0) : z;
  f.p.hi = (k1 < BUF_DIM_) ? *(const us8v*)(row + k1) : z;
  return f.v;
}

// P5 special B: scale_w stored [H=256][D=336]; per-lane contiguous k => 32B load.
__device__ __forceinline__ v16bf load_b_scale(const unsigned short* sw, int ntile,
                                              int kbase, int lane) {
  int n = ntile * 16 + (lane & 15);
  int koff = kbase + ((lane >> 4) << 4);
  BFrag f;
  if (koff < BUF_DIM_) f.u = *(const us16v*)(sw + n * BUF_DIM_ + koff);
  else { us16v z = {}; f.u = z; }
  return f.v;
}

// ---------------- deterministic grid-wide barrier ----------------
__device__ __forceinline__ void grid_barrier(unsigned* bar) {
  __syncthreads();
  if (threadIdx.x == 0) {
    unsigned* cnt = bar;
    unsigned* gen = bar + 1;
    unsigned g = __hip_atomic_load(gen, __ATOMIC_ACQUIRE, __HIP_MEMORY_SCOPE_AGENT);
    __threadfence();
    unsigned v = __hip_atomic_fetch_add(cnt, 1u, __ATOMIC_ACQ_REL,
                                        __HIP_MEMORY_SCOPE_AGENT);
    if (v == (unsigned)(gridDim.x - 1)) {
      __hip_atomic_store(cnt, 0u, __ATOMIC_RELAXED, __HIP_MEMORY_SCOPE_AGENT);
      __hip_atomic_fetch_add(gen, 1u, __ATOMIC_ACQ_REL, __HIP_MEMORY_SCOPE_AGENT);
    } else {
      while (__hip_atomic_load(gen, __ATOMIC_ACQUIRE, __HIP_MEMORY_SCOPE_AGENT) == g)
        __builtin_amdgcn_s_sleep(2);
    }
  }
  __syncthreads();
}

// ---------------- kernel arguments ----------------
struct Args {
  const float *__restrict__ inputs, *__restrict__ targets;
  const float *__restrict__ mp_w1, *__restrict__ mp_b1;
  const float *__restrict__ mp_w2, *__restrict__ mp_b2;
  const float *__restrict__ bp_w1, *__restrict__ bp_b1;
  const float *__restrict__ bp_w2, *__restrict__ bp_b2;
  const float *__restrict__ att_wq, *__restrict__ att_bq;
  const float *__restrict__ att_wa, *__restrict__ att_ba, *__restrict__ att_v;
  const float *__restrict__ scale_w, *__restrict__ scale_b;
  const float *__restrict__ out_w, *__restrict__ out_b;
  float* __restrict__ out;
  // workspace
  unsigned* bar;
  unsigned short *__restrict__ wq_bf, *__restrict__ w1_bf, *__restrict__ w2_bf;
  unsigned short *__restrict__ ow_bf, *__restrict__ sw_bf;
  unsigned short *__restrict__ zbA, *__restrict__ zbB;
  unsigned short *__restrict__ t1, *__restrict__ h_bf;
  float *__restrict__ pa, *__restrict__ mem_h, *__restrict__ mem_p;
  float *__restrict__ part1, *__restrict__ part3, *__restrict__ part6;
};

// fragment-major bf16 weight conversion: dst[(c*ntiles+nt)*512 + lane*16 + j]
__device__ __forceinline__ void conv_frag(unsigned short* dst, const float* src,
                                          int kch, int ntiles, int Kvalid, int Nsrc,
                                          int Nvalid, int gtid) {
  int total = kch * ntiles * 512;
  for (int i = gtid; i < total; i += NTHREADS) {
    int j = i & 15;
    int lane = (i >> 4) & 31;
    int tile = i >> 9;
    int nt = tile % ntiles;
    int c = tile / ntiles;
    int k = c * 32 + ((lane >> 4) << 4) + j;
    int n = nt * 16 + (lane & 15);
    unsigned short v = 0;
    if (k < Kvalid && n < Nvalid) v = f2bf(src[k * Nsrc + n]);
    dst[i] = v;
  }
}

__global__ __launch_bounds__(BLK, 1) void decoder_persistent(Args A) {
  const int tid = threadIdx.x;
  const int gtid = blockIdx.x * BLK + tid;
  const int lane = tid & 31;
  const int wid = gtid >> 5;

  __shared__ float pq_s[HIDDEN_];
  __shared__ float psc[BLK];
  __shared__ float sc[T_ENC_];
  __shared__ float at[T_ENC_];

  // ================= P0a: weight conversion, pa, mem_h, zero buffers =========
  conv_frag(A.wq_bf, A.att_wq, NCH, 16, BUF_SIZE_, HIDDEN_, HIDDEN_, gtid);
  conv_frag(A.w1_bf, A.bp_w1, NCH, 36, BUF_SIZE_, H1V, H1V, gtid);
  conv_frag(A.w2_bf, A.bp_w2, 18, 21, H1V, BUF_DIM_, BUF_DIM_, gtid);
  conv_frag(A.ow_bf, A.out_w, 136, 25, HLK, OUT_DIM_, OUT_DIM_, gtid);
  for (int i = gtid; i < HIDDEN_ * BUF_DIM_; i += NTHREADS)
    A.sw_bf[i] = f2bf(A.scale_w[i]);
  for (int i = gtid; i < B_ * KPAD; i += NTHREADS) { A.zbA[i] = 0; A.zbB[i] = 0; }
  // pa = inputs @ att_wa + ba  (f32, one-time)
  for (int i = gtid; i < B_ * T_ENC_ * HIDDEN_; i += NTHREADS) {
    int row = i >> 8, h = i & 255;
    const float* x = A.inputs + row * IN_DIM_;
    float s = A.att_ba[h];
    for (int k = 0; k < IN_DIM_; ++k) s += x[k] * A.att_wa[k * HIDDEN_ + h];
    A.pa[i] = s;
  }
  // memory-MLP hidden for all t (teacher forcing -> loop-invariant)
  for (int i = gtid; i < T_DEC * B_ * 40; i += NTHREADS) {
    int j = i % 40; int tb = i / 40; int b = tb % B_; int t = tb / B_;
    float s = A.mp_b1[j];
    if (t > 0) {
      const float* mem = A.targets + (b * T_DEC + (t - 1)) * OUT_DIM_;
      for (int k = 0; k < OUT_DIM_; ++k) s += mem[k] * A.mp_w1[k * 40 + j];
    }
    A.mem_h[i] = s > 0.f ? s : 0.f;
  }
  grid_barrier(A.bar);
  // P0b: mem_p for all t
  for (int i = gtid; i < T_DEC * B_ * MEM_DIM_; i += NTHREADS) {
    int j = i % MEM_DIM_; int tb = i / MEM_DIM_;
    float s = A.mp_b2[j];
    const float* hr = A.mem_h + tb * 40;
    for (int k = 0; k < 40; ++k) s += hr[k] * A.mp_w2[k * MEM_DIM_ + j];
    A.mem_p[i] = s;
  }
  grid_barrier(A.bar);

  // ================= sequential decode loop ==================================
  for (int t = 0; t < T_DEC; ++t) {
    const unsigned short* cur = (t & 1) ? A.zbB : A.zbA;   // buf_t (flat bf16)
    unsigned short* nxt = (t & 1) ? A.zbA : A.zbB;         // becomes buf_{t+1}

    // ---- P1: pq partials = buf @ att_wq  (16 ntiles x 16 kgroups = 256 waves)
    {
      int nt = wid & 15, kg = wid >> 4;
      int c0 = kg * 12, c1 = c0 + 12; if (c1 > NCH) c1 = NCH;
      int n = c1 - c0; if (n < 0) n = 0;
      v8f acc = gemm_run(cur + c0 * 32, KPAD,
                         A.wq_bf + (c0 * 16 + nt) * 512, 16 * 512, n, lane);
      store_partial(A.part1 + (kg * 16) * HIDDEN_ + nt * 16, HIDDEN_, acc, lane);
    }
    grid_barrier(A.bar);
    // ---- P2: fused pq reduction + attention + ctx + buffer shift (block b<16)
    if (blockIdx.x < B_) {
      int b = blockIdx.x;
      {  // block-local split-K reduction for pq (consumed only by this block)
        float s = A.att_bq[tid];
        for (int kg = 0; kg < 16; ++kg)
          s += A.part1[(kg * 16 + b) * HIDDEN_ + tid];
        pq_s[tid] = s;
      }
      __syncthreads();
      {  // scores: each (te) dot split across 2 threads
        int te = tid >> 1, half = (tid & 1) << 7;   // 0 or 128
        const float* par = A.pa + (b * T_ENC_ + te) * HIDDEN_ + half;
        const float* pqh = pq_s + half;
        const float* avh = A.att_v + half;
        float s = 0.f;
        for (int h = 0; h < 128; ++h) s += tanhf(pqh[h] + par[h]) * avh[h];
        psc[tid] = s;
      }
      __syncthreads();
      if (tid < T_ENC_) sc[tid] = psc[2 * tid] + psc[2 * tid + 1];
      __syncthreads();
      float mx = -1e30f;
      for (int k = 0; k < T_ENC_; ++k) mx = fmaxf(mx, sc[k]);
      float den = 0.f;
      for (int k = 0; k < T_ENC_; ++k) den += expf(sc[k] - mx);
      float inv = 1.f / den;
      if (tid < T_ENC_) {
        float av = expf(sc[tid] - mx) * inv;
        at[tid] = av;
        A.out[OUTS_ELEMS + (b * T_DEC + t) * T_ENC_ + tid] = av;   // attns out
      }
      __syncthreads();
      {  // ctx[b,i] = sum_te attn * inputs ; write top = [ctx ; mem_p]
        int i = tid;  // 0..255 == IN_DIM
        const float* inb = A.inputs + b * T_ENC_ * IN_DIM_;
        float cv = 0.f;
        for (int te = 0; te < T_ENC_; ++te) cv += at[te] * inb[te * IN_DIM_ + i];
        nxt[b * KPAD + i] = f2bf(cv);
      }
      if (tid < MEM_DIM_)
        nxt[b * KPAD + IN_DIM_ + tid] = f2bf(A.mem_p[(t * B_ + b) * MEM_DIM_ + tid]);
      // shift: rows 1..16 of next buffer <- rows 0..15 of current buffer
      for (int j = tid; j < 16 * BUF_DIM_; j += BLK)
        nxt[b * KPAD + BUF_DIM_ + j] = cur[b * KPAD + j];
    }
    grid_barrier(A.bar);
    // ---- P3: u1 partials = z @ bp_w1 (36 ntiles x 7 kgroups = 252 tasks)
    if (wid < 36 * KG3) {
      int nt = wid % 36, kg = wid / 36;
      int c0 = kg * 26, c1 = c0 + 26; if (c1 > NCH) c1 = NCH;
      v8f acc = gemm_run(nxt + c0 * 32, KPAD,
                         A.w1_bf + (c0 * 36 + nt) * 512, 36 * 512, c1 - c0, lane);
      store_partial(A.part3 + (kg * 16) * H1P + nt * 16, H1P, acc, lane);
    }
    grid_barrier(A.bar);
    // ---- P3b: reduce + bias + relu -> t1 (bf16 16x576)
    for (int i = gtid; i < B_ * H1P; i += NTHREADS) {
      int b = i / H1P, n = i % H1P;
      float s = (n < H1V) ? A.bp_b1[n] : 0.f;
      for (int kg = 0; kg < KG3; ++kg) s += A.part3[(kg * 16 + b) * H1P + n];
      A.t1[i] = f2bf(s > 0.f ? s : 0.f);
    }
    grid_barrier(A.bar);
    // ---- P4: u = t1 @ bp_w2 + b2 -> row 0 of next buffer (21 ntiles)
    if (wid < 21) {
      int nt = wid;
      v8f acc = gemm_run(A.t1, H1P, A.w2_bf + nt * 512, 21 * 512, 18, lane);
      int col = lane & 15, rb = (lane >> 4) << 3;
      int n = nt * 16 + col;
      float bias = A.bp_b2[n];
#pragma unroll
      for (int r = 0; r < 8; ++r)
        nxt[(r + rb) * KPAD + n] = f2bf(acc[r] + bias);
    }
    grid_barrier(A.bar);
    // ---- P5: h = buf_new @ scale_w^T + scale_b  (272x336x256; 17x16 tiles)
    for (int task = wid; task < 17 * 16; task += NWAVES) {
      int mt = task >> 4, ntile = task & 15;
      v8f acc0 = {}, acc1 = {};
      // rotated pipeline over 11 chunks (10 paired + guarded tail)
      v16bf a0 = load_a_h(nxt, mt, 0, lane);
      v16bf b0 = load_b_scale(A.sw_bf, ntile, 0, lane);
      v16bf a1 = load_a_h(nxt, mt, 32, lane);
      v16bf b1 = load_b_scale(A.sw_bf, ntile, 32, lane);
      for (int c = 2; c + 2 <= 10; c += 2) {
        v16bf a2 = load_a_h(nxt, mt, c * 32, lane);
        v16bf b2 = load_b_scale(A.sw_bf, ntile, c * 32, lane);
        v16bf a3 = load_a_h(nxt, mt, (c + 1) * 32, lane);
        v16bf b3 = load_b_scale(A.sw_bf, ntile, (c + 1) * 32, lane);
        acc0 = wmma_bf16(a0, b0, acc0);
        acc1 = wmma_bf16(a1, b1, acc1);
        a0 = a2; b0 = b2; a1 = a3; b1 = b3;
      }
      {  // tail chunk 10 (K 320..335 valid, upper half zero-guarded)
        v16bf a2 = load_a_h(nxt, mt, 320, lane);
        v16bf b2 = load_b_scale(A.sw_bf, ntile, 320, lane);
        acc0 = wmma_bf16(a0, b0, acc0);
        acc1 = wmma_bf16(a1, b1, acc1);
        acc0 = wmma_bf16(a2, b2, acc0);
      }
#pragma unroll
      for (int r = 0; r < 8; ++r) acc0[r] += acc1[r];
      int col = lane & 15, rb = (lane >> 4) << 3;
      int h = ntile * 16 + col;
      float bias = A.scale_b[h];
#pragma unroll
      for (int r = 0; r < 8; ++r) {
        int rg = mt * 16 + r + rb;
        int b = rg / 17, l = rg - b * 17;
        A.h_bf[b * HLK + l * HIDDEN_ + h] = f2bf(acc0[r] + bias);
      }
    }
    grid_barrier(A.bar);
    // ---- P6: out partials = h @ out_w (25 ntiles x 8 kgroups = 200 tasks)
    if (wid < 25 * 8) {
      int nt = wid % 25, kg = wid / 25;
      int c0 = kg * 17;                      // 136 = 8*17 exactly
      v8f acc = gemm_run(A.h_bf + c0 * 32, HLK,
                         A.ow_bf + (c0 * 25 + nt) * 512, 25 * 512, 17, lane);
      store_partial(A.part6 + (kg * 16) * OUT_DIM_ + nt * 16, OUT_DIM_, acc, lane);
    }
    grid_barrier(A.bar);
    // ---- P6b: reduce + bias -> outs[:, t]  (no trailing barrier needed:
    //      next-step P1 only writes part1, which nothing in flight reads)
    for (int i = gtid; i < B_ * OUT_DIM_; i += NTHREADS) {
      int b = i / OUT_DIM_, n = i % OUT_DIM_;
      float s = A.out_b[n];
      for (int kg = 0; kg < 8; ++kg) s += A.part6[(kg * 16 + b) * OUT_DIM_ + n];
      A.out[(b * T_DEC + t) * OUT_DIM_ + n] = s;
    }
  }
}

// ---------------- host launcher ----------------
extern "C" void kernel_launch(void* const* d_in, const int* in_sizes, int n_in,
                              void* d_out, int out_size, void* d_ws, size_t ws_size,
                              hipStream_t stream) {
  (void)in_sizes; (void)n_in; (void)out_size; (void)ws_size;
  Args a;
  a.inputs  = (const float*)d_in[0];
  a.targets = (const float*)d_in[1];
  a.mp_w1 = (const float*)d_in[2];  a.mp_b1 = (const float*)d_in[3];
  a.mp_w2 = (const float*)d_in[4];  a.mp_b2 = (const float*)d_in[5];
  a.bp_w1 = (const float*)d_in[6];  a.bp_b1 = (const float*)d_in[7];
  a.bp_w2 = (const float*)d_in[8];  a.bp_b2 = (const float*)d_in[9];
  a.att_wq = (const float*)d_in[10]; a.att_bq = (const float*)d_in[11];
  a.att_wa = (const float*)d_in[12]; a.att_ba = (const float*)d_in[13];
  a.att_v  = (const float*)d_in[14];
  a.scale_w = (const float*)d_in[15]; a.scale_b = (const float*)d_in[16];
  a.out_w   = (const float*)d_in[17]; a.out_b   = (const float*)d_in[18];
  a.out = (float*)d_out;

  size_t off = 0;
  auto take = [&](size_t bytes) -> void* {
    off = (off + 255) & ~(size_t)255;
    void* r = (char*)d_ws + off;
    off += bytes;
    return r;
  };
  a.bar   = (unsigned*)take(64);
  a.wq_bf = (unsigned short*)take((size_t)NCH * 16 * 512 * 2);
  a.w1_bf = (unsigned short*)take((size_t)NCH * 36 * 512 * 2);
  a.w2_bf = (unsigned short*)take((size_t)18 * 21 * 512 * 2);
  a.ow_bf = (unsigned short*)take((size_t)136 * 25 * 512 * 2);
  a.sw_bf = (unsigned short*)take((size_t)HIDDEN_ * BUF_DIM_ * 2);
  a.zbA   = (unsigned short*)take((size_t)B_ * KPAD * 2);
  a.zbB   = (unsigned short*)take((size_t)B_ * KPAD * 2);
  a.t1    = (unsigned short*)take((size_t)B_ * H1P * 2);
  a.h_bf  = (unsigned short*)take((size_t)B_ * HLK * 2);
  a.pa    = (float*)take((size_t)B_ * T_ENC_ * HIDDEN_ * 4);
  a.mem_h = (float*)take((size_t)T_DEC * B_ * 40 * 4);
  a.mem_p = (float*)take((size_t)T_DEC * B_ * MEM_DIM_ * 4);
  a.part1 = (float*)take((size_t)16 * B_ * HIDDEN_ * 4);
  a.part3 = (float*)take((size_t)KG3 * B_ * H1P * 4);
  a.part6 = (float*)take((size_t)8 * B_ * OUT_DIM_ * 4);

  hipMemsetAsync(a.bar, 0, 64, stream);   // barrier counters must start at 0
  decoder_persistent<<<dim3(NBLK), dim3(BLK), 0, stream>>>(a);
}
